// MessagePassingPC_90692529422975
// MI455X (gfx1250) — compile-verified, matrix operand order checked by reading, not run on hardware
//
#include <hip/hip_runtime.h>
#include <hip/hip_bf16.h>
#include <math.h>

#define N_NODES 4096
#define DFEAT   128
#define N_STEPS 8
#define ETA_PC  0.1f

typedef __bf16 bf16;
typedef __attribute__((ext_vector_type(16))) __bf16 v16bf;
typedef __attribute__((ext_vector_type(8)))  __bf16 v8bf;
typedef __attribute__((ext_vector_type(8)))  float  v8f;

// ---------------------------------------------------------------------------
// WMMA fragment loader.  Both the A operand (Mtx, M x K row-major) and the B
// operand (ActT, N x K row-major == K x N column-major) use the identical
// per-lane layout for V_WMMA_F32_16X16X32_BF16:
//   lane l, r = l&15, hi = l>>4:
//     elements 0..7  = row r, K = k0 + hi*8 .. +7
//     elements 8..15 = row r, K = k0 + 16 + hi*8 .. +7
// i.e. two contiguous 16-byte loads per lane.
// ---------------------------------------------------------------------------
static __device__ __forceinline__ v16bf load_frag(const bf16* __restrict__ base,
                                                  int ld, int row0, int k0, int lane) {
    int r  = lane & 15;
    int hi = lane >> 4;
    const bf16* p = base + (size_t)(row0 + r) * (size_t)ld + k0 + hi * 8;
    v8bf lo = *(const v8bf*)(p);
    v8bf hh = *(const v8bf*)(p + 16);
    return __builtin_shufflevector(lo, hh, 0, 1, 2, 3, 4, 5, 6, 7,
                                           8, 9, 10, 11, 12, 13, 14, 15);
}

static __device__ __forceinline__ v8f wmma_bf16(v16bf a, v16bf b, v8f c) {
    return __builtin_amdgcn_wmma_f32_16x16x32_bf16(false, a, false, b,
                                                   (short)0, c, false, false);
}

// ---------------------------------------------------------------------------
// Register-blocked 32x32 output tile per wave (2x2 fragments).  Per K-chunk
// of 32: 2 A-frags + 2 B-frags (8 x b128 loads) feed 4 WMMAs -> 2 loads per
// WMMA, each operand fragment reused twice in registers.
// ---------------------------------------------------------------------------
struct Acc2x2 { v8f c[2][2]; };

static __device__ __forceinline__ Acc2x2 gemm_tile2x2(const bf16* __restrict__ Mtx,
                                                      const bf16* __restrict__ ActT,
                                                      int m0, int n0, int K, int lane) {
    Acc2x2 acc;
    acc.c[0][0] = (v8f){};
    acc.c[0][1] = (v8f){};
    acc.c[1][0] = (v8f){};
    acc.c[1][1] = (v8f){};
    for (int k = 0; k < K; k += 32) {
        v16bf a0 = load_frag(Mtx,  K, m0,      k, lane);
        v16bf a1 = load_frag(Mtx,  K, m0 + 16, k, lane);
        v16bf b0 = load_frag(ActT, K, n0,      k, lane);
        v16bf b1 = load_frag(ActT, K, n0 + 16, k, lane);
        acc.c[0][0] = wmma_bf16(a0, b0, acc.c[0][0]);
        acc.c[0][1] = wmma_bf16(a0, b1, acc.c[0][1]);
        acc.c[1][0] = wmma_bf16(a1, b0, acc.c[1][0]);
        acc.c[1][1] = wmma_bf16(a1, b1, acc.c[1][1]);
    }
    return acc;
}

// ---------------------------------------------------------------------------
// Init: mu0 = obs where masked else 0; zero the free-energy accumulator.
// ---------------------------------------------------------------------------
__global__ void init_mu_kernel(const float* __restrict__ obs,
                               const unsigned char* __restrict__ mask,
                               float* __restrict__ mu, float* __restrict__ fe) {
    int idx = blockIdx.x * blockDim.x + threadIdx.x;
    if (idx < N_NODES * DFEAT) {
        int i = idx / DFEAT;
        mu[idx] = mask[i] ? obs[idx] : 0.0f;
    }
    if (idx == 0) *fe = 0.0f;
}

// ---------------------------------------------------------------------------
// Convert adjacency int32 -> bf16 (A16 row-major, AT16 = transpose), and
// compute row sums n_chi[i] = sum_j A[i][j].  One block per row.
// ---------------------------------------------------------------------------
__global__ __launch_bounds__(256) void adj_convert_kernel(const int* __restrict__ adj,
                                                          bf16* __restrict__ A16,
                                                          bf16* __restrict__ AT16,
                                                          float* __restrict__ n_chi) {
    __shared__ float sred[256];
    int i = blockIdx.x;
    float s = 0.0f;
    for (int j = threadIdx.x; j < N_NODES; j += 256) {
        float a = (adj[(size_t)i * N_NODES + j] > 0) ? 1.0f : 0.0f;
        A16[(size_t)i * N_NODES + j]  = (bf16)a;
        AT16[(size_t)j * N_NODES + i] = (bf16)a;
        s += a;
    }
    sred[threadIdx.x] = s;
    __syncthreads();
    for (int off = 128; off > 0; off >>= 1) {
        if (threadIdx.x < off) sred[threadIdx.x] += sred[threadIdx.x + off];
        __syncthreads();
    }
    if (threadIdx.x == 0) n_chi[i] = sred[0];
}

// n_par[j] = column sums of A = row sums of AT16 (contiguous reads).
__global__ __launch_bounds__(256) void npar_kernel(const bf16* __restrict__ AT16,
                                                   float* __restrict__ n_par) {
    __shared__ float sred[256];
    int j = blockIdx.x;
    float s = 0.0f;
    for (int i = threadIdx.x; i < N_NODES; i += 256) {
        s += (float)AT16[(size_t)j * N_NODES + i];
    }
    sred[threadIdx.x] = s;
    __syncthreads();
    for (int off = 128; off > 0; off >>= 1) {
        if (threadIdx.x < off) sred[threadIdx.x] += sred[threadIdx.x + off];
        __syncthreads();
    }
    if (threadIdx.x == 0) n_par[j] = sred[0];
}

// ---------------------------------------------------------------------------
// Batched matvec + tanh: pred[n,o] = tanh(sum_d mu[n,d] * W[n,o,d] + b[n,o]).
// Bandwidth-bound on W (268 MB/step; ~92 us/8 steps at 23.3 TB/s HBM).
// One block (128 threads) per node; mu row staged in LDS, W rows read as
// float4.  Output written transposed as bf16 (predT, 128 x 4096) to feed the
// WMMA B operand directly.
// ---------------------------------------------------------------------------
__global__ __launch_bounds__(128) void pred_kernel(const float* __restrict__ mu,
                                                   const float* __restrict__ W,
                                                   const float* __restrict__ b,
                                                   bf16* __restrict__ predT) {
    __shared__ float smu[DFEAT];
    int n = blockIdx.x;
    int o = threadIdx.x;
    smu[o] = mu[(size_t)n * DFEAT + o];
    __syncthreads();
    const float4* wr = (const float4*)(W + ((size_t)n * DFEAT + o) * DFEAT);
    float acc = b[(size_t)n * DFEAT + o];
#pragma unroll 4
    for (int d4 = 0; d4 < DFEAT / 4; ++d4) {
        float4 w = wr[d4];
        acc += smu[d4 * 4 + 0] * w.x + smu[d4 * 4 + 1] * w.y +
               smu[d4 * 4 + 2] * w.z + smu[d4 * 4 + 3] * w.w;
    }
    predT[(size_t)o * N_NODES + n] = (bf16)tanhf(acc);
}

// ---------------------------------------------------------------------------
// GEMM 1: x_pred = (A^T @ pred) / max(n_par,1) gated by n_par>0, fused with
// eps = mu - x_pred (+ (obs-mu) where masked).  256 threads = 8 waves;
// block tile 64 rows x 128 cols, wave w -> rows (w&1)*32, cols (w>>1)*32,
// 2x2 WMMA fragments per wave.  Writes eps f32 (row-major, d_out region)
// and epsT bf16 for GEMM 2.
// ---------------------------------------------------------------------------
__global__ __launch_bounds__(256) void gemm_xpred_eps_kernel(
        const bf16* __restrict__ AT16, const bf16* __restrict__ predT,
        const float* __restrict__ n_par, const float* __restrict__ mu,
        const float* __restrict__ obs, const unsigned char* __restrict__ mask,
        float* __restrict__ eps, bf16* __restrict__ epsT) {
    int lane = threadIdx.x & 31;
    int wave = threadIdx.x >> 5;
    int m0 = blockIdx.x * 64 + (wave & 1) * 32;
    int n0 = (wave >> 1) * 32;
    Acc2x2 acc = gemm_tile2x2(AT16, predT, m0, n0, N_NODES, lane);

    int r = lane & 15, hi = lane >> 4;
#pragma unroll
    for (int mi = 0; mi < 2; ++mi) {
#pragma unroll
        for (int ni = 0; ni < 2; ++ni) {
            int o = n0 + ni * 16 + r;
#pragma unroll
            for (int v = 0; v < 8; ++v) {
                int i = m0 + mi * 16 + hi * 8 + v;
                float np = n_par[i];
                float xp = (np > 0.0f) ? (acc.c[mi][ni][v] / np) : 0.0f;
                size_t idx = (size_t)i * DFEAT + o;
                float m = mu[idx];
                float e = m - xp;
                if (mask[i]) e += obs[idx] - m;
                eps[idx] = e;
                epsT[(size_t)o * N_NODES + i] = (bf16)e;
            }
        }
    }
}

// ---------------------------------------------------------------------------
// GEMM 2: corr = (A @ eps) / max(n_chi,1) gated by n_chi>0, fused with the
// mu update: mu += ETA*(-eps + 0.5*corr), clamped to obs where masked.
// ---------------------------------------------------------------------------
__global__ __launch_bounds__(256) void gemm_corr_mu_kernel(
        const bf16* __restrict__ A16, const bf16* __restrict__ epsT,
        const float* __restrict__ n_chi, const float* __restrict__ eps,
        const float* __restrict__ obs, const unsigned char* __restrict__ mask,
        float* __restrict__ mu) {
    int lane = threadIdx.x & 31;
    int wave = threadIdx.x >> 5;
    int m0 = blockIdx.x * 64 + (wave & 1) * 32;
    int n0 = (wave >> 1) * 32;
    Acc2x2 acc = gemm_tile2x2(A16, epsT, m0, n0, N_NODES, lane);

    int r = lane & 15, hi = lane >> 4;
#pragma unroll
    for (int mi = 0; mi < 2; ++mi) {
#pragma unroll
        for (int ni = 0; ni < 2; ++ni) {
            int o = n0 + ni * 16 + r;
#pragma unroll
            for (int v = 0; v < 8; ++v) {
                int i = m0 + mi * 16 + hi * 8 + v;
                float nch = n_chi[i];
                float cr = (nch > 0.0f) ? (acc.c[mi][ni][v] / nch) : 0.0f;
                size_t idx = (size_t)i * DFEAT + o;
                float e = eps[idx];
                float m = mu[idx];
                float mn = m + ETA_PC * (-e + 0.5f * cr);
                mu[idx] = mask[i] ? obs[idx] : mn;
            }
        }
    }
}

// Free energy = 0.5 * sum(eps^2).
__global__ __launch_bounds__(256) void fe_kernel(const float* __restrict__ eps,
                                                 float* __restrict__ fe) {
    __shared__ float sred[256];
    float s = 0.0f;
    for (int idx = blockIdx.x * 256 + threadIdx.x; idx < N_NODES * DFEAT;
         idx += gridDim.x * 256) {
        float e = eps[idx];
        s += e * e;
    }
    sred[threadIdx.x] = s;
    __syncthreads();
    for (int off = 128; off > 0; off >>= 1) {
        if (threadIdx.x < off) sred[threadIdx.x] += sred[threadIdx.x + off];
        __syncthreads();
    }
    if (threadIdx.x == 0) atomicAdd(fe, 0.5f * sred[0]);
}

extern "C" void kernel_launch(void* const* d_in, const int* in_sizes, int n_in,
                              void* d_out, int out_size, void* d_ws, size_t ws_size,
                              hipStream_t stream) {
    const float*         obs  = (const float*)d_in[0];
    const int*           adj  = (const int*)d_in[1];
    const unsigned char* mask = (const unsigned char*)d_in[2];
    const float*         W    = (const float*)d_in[3];
    const float*         b    = (const float*)d_in[4];

    float* mu  = (float*)d_out;                           // 4096*128
    float* eps = (float*)d_out + N_NODES * DFEAT;         // 4096*128
    float* fe  = (float*)d_out + 2 * N_NODES * DFEAT;     // 1

    char* ws = (char*)d_ws;
    bf16*  A16   = (bf16*)(ws);                 // 32 MB   A  row-major bf16
    bf16*  AT16  = (bf16*)(ws + 33554432);      // 32 MB   A^T row-major bf16
    bf16*  predT = (bf16*)(ws + 67108864);      // 1 MB    pred transposed (128 x 4096)
    bf16*  epsT  = (bf16*)(ws + 68157440);      // 1 MB    eps transposed  (128 x 4096)
    float* n_par = (float*)(ws + 69206016);     // 16 KB
    float* n_chi = (float*)(ws + 69222400);     // 16 KB

    init_mu_kernel<<<(N_NODES * DFEAT + 255) / 256, 256, 0, stream>>>(obs, mask, mu, fe);
    adj_convert_kernel<<<N_NODES, 256, 0, stream>>>(adj, A16, AT16, n_chi);
    npar_kernel<<<N_NODES, 256, 0, stream>>>(AT16, n_par);

    dim3 gemm_grid(N_NODES / 64);
    for (int step = 0; step < N_STEPS; ++step) {
        pred_kernel<<<N_NODES, DFEAT, 0, stream>>>(mu, W, b, predT);
        gemm_xpred_eps_kernel<<<gemm_grid, 256, 0, stream>>>(AT16, predT, n_par,
                                                             mu, obs, mask, eps, epsT);
        gemm_corr_mu_kernel<<<gemm_grid, 256, 0, stream>>>(A16, epsT, n_chi,
                                                           eps, obs, mask, mu);
    }
    fe_kernel<<<512, 256, 0, stream>>>(eps, fe);
}
